// AGAEMD_44427141710332
// MI455X (gfx1250) — matrix operand order repaired and checked by reading, not compiled, not run on hardware
//
#include <hip/hip_runtime.h>

// GAT (2-layer) fused implementation for gfx1250 (CDNA5, wave32, WMMA).
// N=4096, F=256, H=8, D=64, C=64, slope=0.2, mask fill = -9e15 (=> p=0 exactly).

#define GN 4096
#define GF 256
#define GH 8
#define GD 64
#define GC 64
#define GHD 512
#define SLOPE 0.2f

typedef _Float16 h16;
typedef __attribute__((ext_vector_type(16))) _Float16 v16h;
typedef __attribute__((ext_vector_type(8)))  _Float16 v8h;
typedef __attribute__((ext_vector_type(8)))  float    v8f;
typedef int v4i __attribute__((ext_vector_type(4)));

// ---- optional gfx1250 async global->LDS path --------------------------------
#if defined(__has_builtin)
#if __has_builtin(__builtin_amdgcn_global_load_async_to_lds_b128) && \
    __has_builtin(__builtin_amdgcn_s_wait_asynccnt)
#define USE_ASYNC_LDS 1
#endif
#endif

__device__ __forceinline__ void tile_ld128(float4* dst, const float4* src) {
#ifdef USE_ASYNC_LDS
    // signature (probe-derived): (AS1 v4i* gaddr, AS3 v4i* ldsaddr, Ii off, Ii cpol)
    __builtin_amdgcn_global_load_async_to_lds_b128(
        (__attribute__((address_space(1))) v4i*)
            ((__attribute__((address_space(1))) const void*)src),
        (__attribute__((address_space(3))) v4i*)
            ((__attribute__((address_space(3))) void*)dst),
        0, 0);
#else
    *dst = *src;          // global_load_b128 + ds_store_b128 fallback
#endif
}
__device__ __forceinline__ void async_tile_wait() {
#ifdef USE_ASYNC_LDS
    __builtin_amdgcn_s_wait_asynccnt(0);
#endif
}

// ---- WMMA helpers -----------------------------------------------------------

__device__ __forceinline__ v8f wmma_f16(v16h a, v16h b, v8f c) {
    return __builtin_amdgcn_wmma_f32_16x16x32_f16(false, a, false, b,
                                                  (short)0, c, false, false);
}

// A fragment: row-major f16 matrix, p = &M[row_m][k0]. klo = 8*(lane>>4).
__device__ __forceinline__ v16h ldA(const h16* p, int klo) {
    v8h lo = *(const v8h*)(p + 2 * klo);
    v8h hi = *(const v8h*)(p + 2 * klo + 8);
    v16h a;
#pragma unroll
    for (int i = 0; i < 8; ++i) { a[i] = lo[i]; a[8 + i] = hi[i]; }
    return a;
}

// B fragment: K-contiguous transposed layout T[n][k], p = &T[col_n][k0 + kb].
__device__ __forceinline__ v16h ldB(const h16* p) {
    v8h lo = *(const v8h*)(p);
    v8h hi = *(const v8h*)(p + 8);
    v16h b;
#pragma unroll
    for (int i = 0; i < 8; ++i) { b[i] = lo[i]; b[8 + i] = hi[i]; }
    return b;
}

// ---- Kernel 0: fp32 -> f16 conversions --------------------------------------
__global__ void k_convert(const float* __restrict__ x, const float* __restrict__ W,
                          const float* __restrict__ Wo, h16* __restrict__ xh,
                          h16* __restrict__ WT, h16* __restrict__ WoT) {
    int idx = blockIdx.x * blockDim.x + threadIdx.x;
    const int NX = GN * GF, NW = GH * GF * GD, NWO = GHD * GC;
    if (idx < NX) {
        xh[idx] = (h16)x[idx];
    } else if (idx < NX + NW) {
        int t = idx - NX;
        int h = t / (GF * GD), r = t % (GF * GD);
        int f = r / GD, d = r % GD;
        WT[h * GD * GF + d * GF + f] = (h16)W[t];
    } else if (idx < NX + NW + NWO) {
        int t = idx - NX - NW;
        int f = t / GC, c = t % GC;
        WoT[c * GHD + f] = (h16)Wo[t];
    }
}

// ---- Kernel 1: Wh = x @ W[h]  (per-head GEMM, WMMA) -------------------------
__global__ void k_proj1(const h16* __restrict__ xh, const h16* __restrict__ WT,
                        float* __restrict__ Wh32, h16* __restrict__ WhT) {
    int h = blockIdx.x / (GN / 64);
    int i0 = (blockIdx.x % (GN / 64)) * 64;
    int w = threadIdx.x >> 5, lane = threadIdx.x & 31;
    int m = lane & 15, half = lane >> 4;
    int klo = 8 * half, kb = 16 * half, n = m;
    int row0 = i0 + 16 * (w >> 1);
    int dbase = 32 * (w & 1);

    v8f acc0 = {}, acc1 = {};
    const h16* arow = xh + (size_t)(row0 + m) * GF;
    const h16* tbase = WT + (size_t)h * GD * GF;
#pragma unroll
    for (int kc = 0; kc < GF / 32; ++kc) {
        int k0 = kc * 32;
        v16h a  = ldA(arow + k0, klo);
        v16h b0 = ldB(tbase + (size_t)(dbase + n) * GF + k0 + kb);
        v16h b1 = ldB(tbase + (size_t)(dbase + 16 + n) * GF + k0 + kb);
        acc0 = wmma_f16(a, b0, acc0);
        acc1 = wmma_f16(a, b1, acc1);
    }
    float* w32 = Wh32 + (size_t)h * GN * GD;
    h16*   wt  = WhT  + (size_t)h * GD * GN;
#pragma unroll
    for (int r = 0; r < 8; ++r) {
        int row = row0 + r + 8 * half;
        w32[(size_t)row * GD + dbase + n]      = acc0[r];
        w32[(size_t)row * GD + dbase + 16 + n] = acc1[r];
        wt[(size_t)(dbase + n) * GN + row]      = (h16)acc0[r];
        wt[(size_t)(dbase + 16 + n) * GN + row] = (h16)acc1[r];
    }
}

// ---- Kernel 2: f_src/f_dst = Wh . a -----------------------------------------
__global__ void k_fvec(const float* __restrict__ Wh32, const float* __restrict__ a,
                       float* __restrict__ fsrc, float* __restrict__ fdst) {
    int t = blockIdx.x * blockDim.x + threadIdx.x;
    if (t >= GH * GN) return;
    int h = t >> 12, nIdx = t & (GN - 1);
    const float* wr = Wh32 + (size_t)h * GN * GD + (size_t)nIdx * GD;
    const float* ah = a + h * 2 * GD;
    float fs = 0.f, fd = 0.f;
#pragma unroll 8
    for (int d = 0; d < GD; ++d) { float v = wr[d]; fs += v * ah[d]; fd += v * ah[GD + d]; }
    fsrc[t] = fs; fdst[t] = fd;
}

// ---- Kernel 3: fused masked softmax(attn) @ Wh + ELU -> x2 (f16) ------------
// 32 rows/block, 256 threads = 8 waves (wave w = head w).
__global__ void k_attn1(const float* __restrict__ adj, const float* __restrict__ fsrc,
                        const float* __restrict__ fdst, const h16* __restrict__ WhT,
                        h16* __restrict__ x2) {
    __shared__ __align__(16) float adjS[32 * 32];
    __shared__ float fdS[GH * 32];
    int i0 = blockIdx.x * 32;
    int h = threadIdx.x >> 5, lane = threadIdx.x & 31;
    int m = lane & 15, half = lane >> 4;
    int klo = 8 * half, kb = 16 * half, n = m;
    int t = threadIdx.x;
    int lr = t >> 3, lc = (t & 7) * 4;          // cooperative float4 tile coords

    float fs0 = fsrc[(size_t)h * GN + i0 + m];
    float fs1 = fsrc[(size_t)h * GN + i0 + 16 + m];
    const h16* wtb = WhT + (size_t)h * GD * GN;
    v8f acc0[4] = {{}, {}, {}, {}};
    v8f acc1[4] = {{}, {}, {}, {}};
    float dsum0 = 0.f, dsum1 = 0.f;

    for (int jc = 0; jc < GN / 32; ++jc) {
        int jbase = jc * 32;
        const float* gsrcp = adj + (size_t)(i0 + lr) * GN + jbase + lc;
        tile_ld128((float4*)&adjS[lr * 32 + lc], (const float4*)gsrcp);
        fdS[t] = fdst[(size_t)(t >> 5) * GN + jbase + (t & 31)];
        if (jc + 1 < GN / 32) __builtin_prefetch(gsrcp + 32, 0, 0);
        async_tile_wait();
        __syncthreads();

        v16h pa0, pa1;
#pragma unroll
        for (int i = 0; i < 16; ++i) {
            int jj = kb + i;
            float fd = fdS[h * 32 + jj];
            float a0 = adjS[m * 32 + jj];
            float a1 = adjS[(16 + m) * 32 + jj];
            float v0 = fs0 + fd; v0 = (v0 > 0.f) ? v0 : SLOPE * v0;
            float v1 = fs1 + fd; v1 = (v1 > 0.f) ? v1 : SLOPE * v1;
            float p0 = (a0 > 0.f) ? __expf(v0) : 0.f;
            float p1 = (a1 > 0.f) ? __expf(v1) : 0.f;
            dsum0 += p0; dsum1 += p1;
            pa0[i] = (h16)p0; pa1[i] = (h16)p1;
        }
        size_t jo = (size_t)jbase + kb;
#pragma unroll
        for (int dt = 0; dt < 4; ++dt) {
            v16h b = ldB(wtb + (size_t)(dt * 16 + n) * GN + jo);
            acc0[dt] = wmma_f16(pa0, b, acc0[dt]);
            acc1[dt] = wmma_f16(pa1, b, acc1[dt]);
        }
        __syncthreads();
    }
    float dfull0 = dsum0 + __shfl_xor(dsum0, 16);
    float dfull1 = dsum1 + __shfl_xor(dsum1, 16);
#pragma unroll
    for (int rt = 0; rt < 2; ++rt) {
        float dfull = rt ? dfull1 : dfull0;
        v8f* acc = rt ? acc1 : acc0;
#pragma unroll
        for (int r = 0; r < 8; ++r) {
            float den = __shfl(dfull, r + 8 * half);
            int row = i0 + 16 * rt + r + 8 * half;
            h16* xr = x2 + (size_t)row * GHD + h * GD + n;
#pragma unroll
            for (int dt = 0; dt < 4; ++dt) {
                float v = acc[dt][r] / den;
                v = (v > 0.f) ? v : (__expf(v) - 1.f);   // ELU
                xr[dt * 16] = (h16)v;
            }
        }
    }
}

// ---- Kernel 4: Who = x2 @ Wo (WMMA) -----------------------------------------
__global__ void k_proj2(const h16* __restrict__ x2, const h16* __restrict__ WoT,
                        float* __restrict__ Who32, h16* __restrict__ WhoT) {
    int w = threadIdx.x >> 5, lane = threadIdx.x & 31;
    int m = lane & 15, half = lane >> 4;
    int klo = 8 * half, kb = 16 * half, n = m;
    int i0 = blockIdx.x * 128 + w * 16;

    v8f acc[4] = {{}, {}, {}, {}};
    const h16* arow = x2 + (size_t)(i0 + m) * GHD;
#pragma unroll
    for (int kc = 0; kc < GHD / 32; ++kc) {
        int k0 = kc * 32;
        v16h a = ldA(arow + k0, klo);
#pragma unroll
        for (int ct = 0; ct < 4; ++ct) {
            v16h b = ldB(WoT + (size_t)(ct * 16 + n) * GHD + k0 + kb);
            acc[ct] = wmma_f16(a, b, acc[ct]);
        }
    }
#pragma unroll
    for (int ct = 0; ct < 4; ++ct) {
#pragma unroll
        for (int r = 0; r < 8; ++r) {
            int row = i0 + r + 8 * half;
            int c = ct * 16 + n;
            Who32[(size_t)row * GC + c] = acc[ct][r];
            WhoT[(size_t)c * GN + row] = (h16)acc[ct][r];
        }
    }
}

// ---- Kernel 5: g_src/g_dst = Who . ao ---------------------------------------
__global__ void k_gvec(const float* __restrict__ Who32, const float* __restrict__ ao,
                       float* __restrict__ gsrc, float* __restrict__ gdst) {
    int nIdx = blockIdx.x * blockDim.x + threadIdx.x;
    if (nIdx >= GN) return;
    const float* wr = Who32 + (size_t)nIdx * GC;
    float gs = 0.f, gd = 0.f;
#pragma unroll 8
    for (int c = 0; c < GC; ++c) { float v = wr[c]; gs += v * ao[c]; gd += v * ao[GC + c]; }
    gsrc[nIdx] = gs; gdst[nIdx] = gd;
}

// ---- Kernel 6: layer-2 attention + ELU + row log_softmax -> d_out -----------
// 32 rows/block, 128 threads = 4 waves; wave w owns output cols 16*w..16*w+15.
__global__ void k_attn2(const float* __restrict__ adj, const float* __restrict__ gsrc,
                        const float* __restrict__ gdst, const h16* __restrict__ WhoT,
                        float* __restrict__ out) {
    __shared__ __align__(16) float adjS[32 * 32];
    __shared__ float gdS[32];
    __shared__ float outS[32 * GC];
    int i0 = blockIdx.x * 32;
    int w = threadIdx.x >> 5, lane = threadIdx.x & 31;
    int m = lane & 15, half = lane >> 4;
    int klo = 8 * half, kb = 16 * half, n = m;
    int dbase = w * 16;
    int t = threadIdx.x;

    float gs0 = gsrc[i0 + m];
    float gs1 = gsrc[i0 + 16 + m];
    v8f acc0 = {}, acc1 = {};
    float dsum0 = 0.f, dsum1 = 0.f;

    for (int jc = 0; jc < GN / 32; ++jc) {
        int jbase = jc * 32;
#pragma unroll
        for (int e4 = 0; e4 < 2; ++e4) {
            int e = t + e4 * 128;
            int lr = e >> 3, lc = (e & 7) * 4;
            const float* gsrcp = adj + (size_t)(i0 + lr) * GN + jbase + lc;
            tile_ld128((float4*)&adjS[lr * 32 + lc], (const float4*)gsrcp);
            if (jc + 1 < GN / 32) __builtin_prefetch(gsrcp + 32, 0, 0);
        }
        if (t < 32) gdS[t] = gdst[jbase + t];
        async_tile_wait();
        __syncthreads();

        v16h pa0, pa1;
#pragma unroll
        for (int i = 0; i < 16; ++i) {
            int jj = kb + i;
            float gd = gdS[jj];
            float a0 = adjS[m * 32 + jj];
            float a1 = adjS[(16 + m) * 32 + jj];
            float v0 = gs0 + gd; v0 = (v0 > 0.f) ? v0 : SLOPE * v0;
            float v1 = gs1 + gd; v1 = (v1 > 0.f) ? v1 : SLOPE * v1;
            float p0 = (a0 > 0.f) ? __expf(v0) : 0.f;
            float p1 = (a1 > 0.f) ? __expf(v1) : 0.f;
            dsum0 += p0; dsum1 += p1;
            pa0[i] = (h16)p0; pa1[i] = (h16)p1;
        }
        v16h b = ldB(WhoT + (size_t)(dbase + n) * GN + jbase + kb);
        acc0 = wmma_f16(pa0, b, acc0);
        acc1 = wmma_f16(pa1, b, acc1);
        __syncthreads();
    }
    float dfull0 = dsum0 + __shfl_xor(dsum0, 16);
    float dfull1 = dsum1 + __shfl_xor(dsum1, 16);
#pragma unroll
    for (int rt = 0; rt < 2; ++rt) {
        float dfull = rt ? dfull1 : dfull0;
        v8f acc = rt ? acc1 : acc0;
#pragma unroll
        for (int r = 0; r < 8; ++r) {
            float den = __shfl(dfull, r + 8 * half);
            float v = acc[r] / den;
            v = (v > 0.f) ? v : (__expf(v) - 1.f);   // ELU
            outS[(16 * rt + r + 8 * half) * GC + dbase + n] = v;
        }
    }
    __syncthreads();
    if (t < 32) {
        const float* rp = outS + t * GC;
        float mx = rp[0];
        for (int c = 1; c < GC; ++c) mx = fmaxf(mx, rp[c]);
        float s = 0.f;
        for (int c = 0; c < GC; ++c) s += __expf(rp[c] - mx);
        float lse = mx + __logf(s);
        float* op = out + (size_t)(i0 + t) * GC;
        for (int c = 0; c < GC; ++c) op[c] = rp[c] - lse;
    }
}

// ---- Launch -----------------------------------------------------------------

extern "C" void kernel_launch(void* const* d_in, const int* in_sizes, int n_in,
                              void* d_out, int out_size, void* d_ws, size_t ws_size,
                              hipStream_t stream) {
    const float* x   = (const float*)d_in[0];
    const float* adj = (const float*)d_in[1];
    const float* W   = (const float*)d_in[2];
    const float* a   = (const float*)d_in[3];
    const float* Wo  = (const float*)d_in[4];
    const float* ao  = (const float*)d_in[5];
    float* out = (float*)d_out;

    char* ws = (char*)d_ws;
    size_t o = 0;
    h16* xh = (h16*)(ws + o);     o += (size_t)GN * GF * 2;
    h16* WT = (h16*)(ws + o);     o += (size_t)GH * GD * GF * 2;
    h16* WoT = (h16*)(ws + o);    o += (size_t)GC * GHD * 2;
    float* Wh32 = (float*)(ws + o); o += (size_t)GH * GN * GD * 4;
    h16* WhT = (h16*)(ws + o);    o += (size_t)GH * GD * GN * 2;
    float* fsrc = (float*)(ws + o); o += (size_t)GH * GN * 4;
    float* fdst = (float*)(ws + o); o += (size_t)GH * GN * 4;
    h16* x2 = (h16*)(ws + o);     o += (size_t)GN * GHD * 2;
    float* Who32 = (float*)(ws + o); o += (size_t)GN * GC * 4;
    h16* WhoT = (h16*)(ws + o);   o += (size_t)GC * GN * 2;
    float* gsrc = (float*)(ws + o); o += (size_t)GN * 4;
    float* gdst = (float*)(ws + o); o += (size_t)GN * 4;

    {
        int total = GN * GF + GH * GF * GD + GHD * GC;
        k_convert<<<(total + 255) / 256, 256, 0, stream>>>(x, W, Wo, xh, WT, WoT);
    }
    k_proj1<<<GH * (GN / 64), 256, 0, stream>>>(xh, WT, Wh32, WhT);
    k_fvec<<<(GH * GN) / 256, 256, 0, stream>>>(Wh32, a, fsrc, fdst);
    k_attn1<<<GN / 32, 256, 0, stream>>>(adj, fsrc, fdst, WhT, x2);
    k_proj2<<<GN / 128, 256, 0, stream>>>(x2, WoT, Who32, WhoT);
    k_gvec<<<(GN + 255) / 256, 256, 0, stream>>>(Who32, ao, gsrc, gdst);
    k_attn2<<<GN / 32, 128, 0, stream>>>(adj, gsrc, gdst, WhoT, out);
}
